// TemporalDiscriminator_86990267613751
// MI455X (gfx1250) — matrix operand order repaired and checked by reading, not compiled, not run on hardware
//
#include <hip/hip_runtime.h>
#include <hip/hip_bf16.h>
#include <math.h>

typedef __attribute__((ext_vector_type(16))) _Float16 v16h;
typedef __attribute__((ext_vector_type(8)))  float    v8f;

// ---------------------------------------------------------------------------
// WMMA GEMM:  C[M,N] = A[M,K](f16) * Bt[N,K](f16)^T  (+ bias[N])
// grid = (M/32, N/512), block = 256 (8 waves). Each wave: 32x64 output tile.
// Software-pipelined: two k-chunks in flight (ping-pong register buffers).
// K multiple of 64-compatible (K/32 even); N multiple of 512; M multiple of 32.
// ---------------------------------------------------------------------------
template <typename OutT>
__global__ __launch_bounds__(256) void wmma_gemm_kernel(
    const _Float16* __restrict__ A,   // [M,K] row-major
    const _Float16* __restrict__ Bt,  // [N,K] row-major (i.e. B transposed)
    OutT* __restrict__ C,             // [M,N]
    const float* __restrict__ bias,   // [N] or nullptr
    int M, int N, int K)
{
    const int lane = threadIdx.x & 31;
    const int wave = threadIdx.x >> 5;
    const int row32   = blockIdx.x * 32;
    const int colBase = (blockIdx.y * 8 + wave) * 64;
    const int r  = lane & 15;
    const int hi = lane >> 4;

    const v8f zero = {0.f, 0.f, 0.f, 0.f, 0.f, 0.f, 0.f, 0.f};
    v8f acc[2][4] = {{zero, zero, zero, zero}, {zero, zero, zero, zero}};

    // A fragment: lane holds row, K elems kb..kb+7 and kb+16..kb+23, kb=hi*8
    const _Float16* aRow0 = A + (size_t)(row32 + r) * K + hi * 8;
    const _Float16* aRow1 = A + (size_t)(row32 + 16 + r) * K + hi * 8;
    // B fragment: lane holds col, K elems kb..kb+7 and kb+16..kb+23, kb=hi*8
    const _Float16* bCol  = Bt + (size_t)(colBase + r) * K + hi * 8;

    auto loadA = [](const _Float16* p) -> v16h {
        v16h f;
        ((float4*)&f)[0] = *(const float4*)(p);
        ((float4*)&f)[1] = *(const float4*)(p + 16);
        return f;
    };
    auto loadB = [](const _Float16* p) -> v16h {
        v16h f;
        ((float4*)&f)[0] = *(const float4*)(p);
        ((float4*)&f)[1] = *(const float4*)(p + 16);
        return f;
    };

    // preload chunk 0 into buffer P
    v16h aP0 = loadA(aRow0), aP1 = loadA(aRow1);
    v16h bP[4];
#pragma unroll
    for (int j = 0; j < 4; ++j) bP[j] = loadB(bCol + (size_t)j * 16 * K);

    v16h aQ0, aQ1, bQ[4];

    for (int kc = 0; kc < K; kc += 64) {
        // prefetch chunk kc+32 into buffer Q (always valid: kc+32 <= K-32)
        const int k1 = kc + 32;
        aQ0 = loadA(aRow0 + k1);
        aQ1 = loadA(aRow1 + k1);
#pragma unroll
        for (int j = 0; j < 4; ++j) bQ[j] = loadB(bCol + (size_t)j * 16 * K + k1);

        // compute on buffer P (chunk kc)
#pragma unroll
        for (int j = 0; j < 4; ++j) {
            acc[0][j] = __builtin_amdgcn_wmma_f32_16x16x32_f16(
                false, aP0, false, bP[j], (short)0, acc[0][j], false, false);
            acc[1][j] = __builtin_amdgcn_wmma_f32_16x16x32_f16(
                false, aP1, false, bP[j], (short)0, acc[1][j], false, false);
        }

        // prefetch chunk kc+64 into buffer P (clamped in-bounds; unused on last iter)
        int k2 = kc + 64; if (k2 > K - 32) k2 = K - 32;
        aP0 = loadA(aRow0 + k2);
        aP1 = loadA(aRow1 + k2);
#pragma unroll
        for (int j = 0; j < 4; ++j) bP[j] = loadB(bCol + (size_t)j * 16 * K + k2);

        // compute on buffer Q (chunk kc+32)
#pragma unroll
        for (int j = 0; j < 4; ++j) {
            acc[0][j] = __builtin_amdgcn_wmma_f32_16x16x32_f16(
                false, aQ0, false, bQ[j], (short)0, acc[0][j], false, false);
            acc[1][j] = __builtin_amdgcn_wmma_f32_16x16x32_f16(
                false, aQ1, false, bQ[j], (short)0, acc[1][j], false, false);
        }
    }

#pragma unroll
    for (int half = 0; half < 2; ++half) {
#pragma unroll
        for (int j = 0; j < 4; ++j) {
            int col = colBase + j * 16 + r;
            float bv = bias ? bias[col] : 0.0f;
#pragma unroll
            for (int v = 0; v < 8; ++v) {
                int row = row32 + half * 16 + v + 8 * hi;  // C/D: VGPR v -> M = v+8*hi
                C[(size_t)row * N + col] = (OutT)(acc[half][j][v] + bv);
            }
        }
    }
}

// ---------------------------------------------------------------------------
// conv weight pack: conv_w[D=512][C=3][KT=3][8][8] -> Wct[d][k], k=(kt*3+c)*64+p
// ---------------------------------------------------------------------------
__global__ void pack_convw_kernel(const float* __restrict__ w, _Float16* __restrict__ out)
{
    int gid = blockIdx.x * blockDim.x + threadIdx.x;
    if (gid >= 512 * 576) return;
    int k = gid % 576, d = gid / 576;
    int kt = k / 192, rem = k % 192, c = rem / 64, p = rem % 64;
    out[gid] = (_Float16)w[((d * 3 + c) * 3 + kt) * 64 + p];
}

// ---------------------------------------------------------------------------
// im2col with causal temporal zero-pad:
// row m = ((b*32+t)*8+hn)*8+wn ; col k = (kt*3+c)*64 + ph*8 + pw ; tsrc=t+kt-2
// ---------------------------------------------------------------------------
__global__ void im2col_kernel(const float* __restrict__ x, _Float16* __restrict__ A)
{
    size_t gid = (size_t)blockIdx.x * blockDim.x + threadIdx.x;
    if (gid >= (size_t)32768 * 576) return;
    int k = (int)(gid % 576);
    int m = (int)(gid / 576);
    int wn = m & 7, hn = (m >> 3) & 7;
    int t = (m >> 6) & 31, b = m >> 11;
    int kt = k / 192, rem = k % 192;
    int c = rem / 64, p = rem % 64;
    int ph = p >> 3, pw = p & 7;
    int ts = t + kt - 2;
    float val = 0.0f;
    if (ts >= 0)
        val = x[(((size_t)(b * 32 + ts) * 3 + c) * 64 + (hn * 8 + ph)) * 64 + (wn * 8 + pw)];
    A[gid] = (_Float16)val;
}

// ---------------------------------------------------------------------------
// generic weight transpose pack: in[K][N] f32 -> out[N][K] f16
// ---------------------------------------------------------------------------
__global__ void pack_transpose_kernel(const float* __restrict__ in, _Float16* __restrict__ out,
                                      int K, int N)
{
    int gid = blockIdx.x * blockDim.x + threadIdx.x;
    if (gid >= K * N) return;
    int n = gid % N, k = gid / N;
    out[(size_t)n * K + k] = (_Float16)in[gid];
}

// ---------------------------------------------------------------------------
// FiLM: gb[r][j] = z[r] . film_w[:,j] + film_b[j]   (r<512, j<1024, K=32)
// ---------------------------------------------------------------------------
__global__ void film_kernel(const float* __restrict__ z, const float* __restrict__ fw,
                            const float* __restrict__ fb, float* __restrict__ gb)
{
    int gid = blockIdx.x * blockDim.x + threadIdx.x;
    if (gid >= 512 * 1024) return;
    int j = gid % 1024, r = gid / 1024;
    float acc = fb[j];
#pragma unroll
    for (int k = 0; k < 32; ++k) acc += z[r * 32 + k] * fw[k * 1024 + j];
    gb[gid] = acc;
}

// ---------------------------------------------------------------------------
// tokenizer tail: +conv_b, LayerNorm(D), FiLM, mean over 64 spatial tokens
// one block per (b,t); 256 threads own 2 channels each
// ---------------------------------------------------------------------------
__global__ __launch_bounds__(256) void tokenize_kernel(
    const _Float16* __restrict__ tok, const float* __restrict__ conv_b,
    const float* __restrict__ tg, const float* __restrict__ tb,
    const float* __restrict__ gb, float* __restrict__ h)
{
    __shared__ float red[256];
    int r = blockIdx.x, tid = threadIdx.x;
    int d0 = tid, d1 = tid + 256;
    float cb0 = conv_b[d0], cb1 = conv_b[d1];
    float ga0 = gb[(size_t)r * 1024 + d0], ga1 = gb[(size_t)r * 1024 + d1];
    float be0 = gb[(size_t)r * 1024 + 512 + d0], be1 = gb[(size_t)r * 1024 + 512 + d1];
    float w0 = tg[d0], w1 = tg[d1], bb0 = tb[d0], bb1 = tb[d1];
    float acc0 = 0.f, acc1 = 0.f;
    for (int n = 0; n < 64; ++n) {
        const _Float16* row = tok + ((size_t)r * 64 + n) * 512;
        float x0 = (float)row[d0] + cb0;
        float x1 = (float)row[d1] + cb1;
        red[tid] = x0 + x1; __syncthreads();
        for (int s = 128; s > 0; s >>= 1) { if (tid < s) red[tid] += red[tid + s]; __syncthreads(); }
        float mu = red[0] * (1.0f / 512.0f); __syncthreads();
        red[tid] = x0 * x0 + x1 * x1; __syncthreads();
        for (int s = 128; s > 0; s >>= 1) { if (tid < s) red[tid] += red[tid + s]; __syncthreads(); }
        float var = red[0] * (1.0f / 512.0f) - mu * mu; __syncthreads();
        float inv = rsqrtf(var + 1e-5f);
        float l0 = (x0 - mu) * inv * w0 + bb0;
        float l1 = (x1 - mu) * inv * w1 + bb1;
        acc0 += (1.0f + 0.5f * ga0) * l0 + 0.5f * be0;
        acc1 += (1.0f + 0.5f * ga1) * l1 + 0.5f * be1;
    }
    h[(size_t)r * 512 + d0] = acc0 * (1.0f / 64.0f);
    h[(size_t)r * 512 + d1] = acc1 * (1.0f / 64.0f);
}

// ---------------------------------------------------------------------------
// LayerNorm rows of h[512][512] -> f16 GEMM operand
// ---------------------------------------------------------------------------
__global__ __launch_bounds__(256) void ln_pack_kernel(
    const float* __restrict__ in, const float* __restrict__ g, const float* __restrict__ b,
    _Float16* __restrict__ out)
{
    __shared__ float red[256];
    int r = blockIdx.x, tid = threadIdx.x;
    float x0 = in[(size_t)r * 512 + tid], x1 = in[(size_t)r * 512 + tid + 256];
    red[tid] = x0 + x1; __syncthreads();
    for (int s = 128; s > 0; s >>= 1) { if (tid < s) red[tid] += red[tid + s]; __syncthreads(); }
    float mu = red[0] * (1.0f / 512.0f); __syncthreads();
    red[tid] = x0 * x0 + x1 * x1; __syncthreads();
    for (int s = 128; s > 0; s >>= 1) { if (tid < s) red[tid] += red[tid + s]; __syncthreads(); }
    float var = red[0] * (1.0f / 512.0f) - mu * mu;
    float inv = rsqrtf(var + 1e-5f);
    out[(size_t)r * 512 + tid]       = (_Float16)((x0 - mu) * inv * g[tid] + b[tid]);
    out[(size_t)r * 512 + tid + 256] = (_Float16)((x1 - mu) * inv * g[tid + 256] + b[tid + 256]);
}

// ---------------------------------------------------------------------------
// attention: one block per (b, head). kqv[512][1536] fp32 (chunk order k,q,v),
// RoPE over time, causal softmax, outputs Y[512][512] f16 (proj GEMM operand)
// ---------------------------------------------------------------------------
__global__ __launch_bounds__(256) void attention_kernel(
    const float* __restrict__ kqv, _Float16* __restrict__ Y)
{
    __shared__ float qs[32 * 64], ks[32 * 64], vs[32 * 64], att[32 * 32];
    int b = blockIdx.x >> 3, hd = blockIdx.x & 7;
    int tid = threadIdx.x;

    for (int idx = tid; idx < 2048; idx += 256) {
        int t = idx >> 6, dd = idx & 63;
        size_t base = (size_t)(b * 32 + t) * 1536 + hd * 64;
        float kr = kqv[base + dd];
        float qr = kqv[base + 512 + dd];
        float vr = kqv[base + 1024 + dd];
        float kp = kqv[base + (dd ^ 32)];
        float qp = kqv[base + 512 + (dd ^ 32)];
        int j = dd & 31;
        float invf = powf(10000.0f, -(float)j * (1.0f / 32.0f));
        float ang = (float)t * invf;
        float cs = cosf(ang), sn = sinf(ang);
        float qrot = (dd < 32) ? -qp : qp;
        float krot = (dd < 32) ? -kp : kp;
        qs[idx] = qr * cs + qrot * sn;
        ks[idx] = kr * cs + krot * sn;
        vs[idx] = vr;
    }
    __syncthreads();
    for (int idx = tid; idx < 1024; idx += 256) {
        int tq = idx >> 5, tk = idx & 31;
        float s;
        if (tk <= tq) {
            float acc = 0.f;
            for (int d = 0; d < 64; ++d) acc += qs[tq * 64 + d] * ks[tk * 64 + d];
            s = acc * 0.125f;   // 1/sqrt(64)
        } else {
            s = -3.0e38f;
        }
        att[idx] = s;
    }
    __syncthreads();
    if (tid < 32) {
        float m = att[tid * 32];
        for (int t2 = 1; t2 < 32; ++t2) m = fmaxf(m, att[tid * 32 + t2]);
        float sum = 0.f;
        for (int t2 = 0; t2 < 32; ++t2) { float e = expf(att[tid * 32 + t2] - m); att[tid * 32 + t2] = e; sum += e; }
        float inv = 1.0f / sum;
        for (int t2 = 0; t2 < 32; ++t2) att[tid * 32 + t2] *= inv;
    }
    __syncthreads();
    for (int idx = tid; idx < 2048; idx += 256) {
        int tq = idx >> 6, dd = idx & 63;
        float acc = 0.f;
        for (int tk = 0; tk < 32; ++tk) acc += att[tq * 32 + tk] * vs[tk * 64 + dd];
        Y[(size_t)(b * 32 + tq) * 512 + hd * 64 + dd] = (_Float16)acc;
    }
}

// ---------------------------------------------------------------------------
__global__ void residual_kernel(float* __restrict__ h, const float* __restrict__ c,
                                const float* __restrict__ rs)
{
    int gid = blockIdx.x * blockDim.x + threadIdx.x;
    if (gid >= 512 * 512) return;
    h[gid] += (*rs) * c[gid];
}

__global__ void gelu_kernel(const float* __restrict__ in, _Float16* __restrict__ out, int n)
{
    int gid = blockIdx.x * blockDim.x + threadIdx.x;
    if (gid >= n) return;
    float x = in[gid];
    out[gid] = (_Float16)(0.5f * x * (1.0f + erff(x * 0.70710678118654752f)));
}

// ---------------------------------------------------------------------------
// head: LN(h) . head_w + head_bias -> out[b*32+t]
// ---------------------------------------------------------------------------
__global__ __launch_bounds__(256) void head_kernel(
    const float* __restrict__ h, const float* __restrict__ g, const float* __restrict__ b,
    const float* __restrict__ w, const float* __restrict__ bias, float* __restrict__ out)
{
    __shared__ float red[256];
    int r = blockIdx.x, tid = threadIdx.x;
    float x0 = h[(size_t)r * 512 + tid], x1 = h[(size_t)r * 512 + tid + 256];
    red[tid] = x0 + x1; __syncthreads();
    for (int s = 128; s > 0; s >>= 1) { if (tid < s) red[tid] += red[tid + s]; __syncthreads(); }
    float mu = red[0] * (1.0f / 512.0f); __syncthreads();
    red[tid] = x0 * x0 + x1 * x1; __syncthreads();
    for (int s = 128; s > 0; s >>= 1) { if (tid < s) red[tid] += red[tid + s]; __syncthreads(); }
    float var = red[0] * (1.0f / 512.0f) - mu * mu; __syncthreads();
    float inv = rsqrtf(var + 1e-5f);
    float l0 = (x0 - mu) * inv * g[tid] + b[tid];
    float l1 = (x1 - mu) * inv * g[tid + 256] + b[tid + 256];
    red[tid] = l0 * w[tid] + l1 * w[tid + 256]; __syncthreads();
    for (int s = 128; s > 0; s >>= 1) { if (tid < s) red[tid] += red[tid + s]; __syncthreads(); }
    if (tid == 0) out[r] = red[0] + bias[0];
}

// ---------------------------------------------------------------------------
extern "C" void kernel_launch(void* const* d_in, const int* in_sizes, int n_in,
                              void* d_out, int out_size, void* d_ws, size_t ws_size,
                              hipStream_t stream)
{
    (void)in_sizes; (void)n_in; (void)out_size; (void)ws_size;
    const float* x       = (const float*)d_in[0];
    const float* z       = (const float*)d_in[1];
    const float* conv_w  = (const float*)d_in[2];
    const float* conv_b  = (const float*)d_in[3];
    const float* tok_g   = (const float*)d_in[4];
    const float* tok_b   = (const float*)d_in[5];
    const float* film_w  = (const float*)d_in[6];
    const float* film_b  = (const float*)d_in[7];
    const float* ln1_g   = (const float*)d_in[8];
    const float* ln1_b   = (const float*)d_in[9];
    const float* kqv_w   = (const float*)d_in[10];
    const float* kqv_b   = (const float*)d_in[11];
    const float* proj_w  = (const float*)d_in[12];
    const float* proj_b  = (const float*)d_in[13];
    const float* ln2_g   = (const float*)d_in[14];
    const float* ln2_b   = (const float*)d_in[15];
    const float* mlp_w1  = (const float*)d_in[16];
    const float* mlp_b1  = (const float*)d_in[17];
    const float* mlp_w2  = (const float*)d_in[18];
    const float* mlp_b2  = (const float*)d_in[19];
    const float* rs_attn = (const float*)d_in[20];
    const float* rs_mlp  = (const float*)d_in[21];
    const float* head_g  = (const float*)d_in[22];
    const float* head_b  = (const float*)d_in[23];
    const float* head_w  = (const float*)d_in[24];
    const float* head_bs = (const float*)d_in[25];

    char* ws = (char*)d_ws;
    // persistent buffers
    float* h  = (float*)(ws);                     // 512*512*4   = 1 MiB
    float* gb = (float*)(ws + (1u << 20));        // 512*1024*4  = 2 MiB
    char* pool = ws + (3u << 20);
    // ---- phase A (tokenizer) view of pool ----
    _Float16* Aim  = (_Float16*)(pool);                         // 32768*576*2 = 37,748,736
    _Float16* Wct  = (_Float16*)(pool + 37748736);              //   512*576*2 =    589,824
    _Float16* tokh = (_Float16*)(pool + 37748736 + 589824);     // 32768*512*2 = 33,554,432
    // ---- phase B (transformer) view of pool (reuses phase-A space) ----
    _Float16* kqvT  = (_Float16*)(pool);                                   // 9,437,184
    _Float16* projT = (_Float16*)(pool + 9437184);                         // 3,145,728
    _Float16* m1T   = (_Float16*)(pool + 9437184 + 3145728);               // 12,582,912
    _Float16* m2T   = (_Float16*)(pool + 9437184 + 3145728 + 12582912);    // 12,582,912
    char* buf = pool + 37748736;
    _Float16* xi    = (_Float16*)(buf);                                    //   524,288
    float*    kqv   = (float*)(buf + 524288);                              // 3,145,728
    _Float16* Yh    = (_Float16*)(buf + 524288 + 3145728);                 //   524,288
    float*    attnC = (float*)(buf + 524288 + 3145728 + 524288);           // 1,048,576
    float*    m1    = (float*)(buf + 524288 + 3145728 + 524288 + 1048576); // 4,194,304
    _Float16* g16   = (_Float16*)((char*)m1 + 4194304);                    // 2,097,152
    float*    m2    = (float*)((char*)g16 + 2097152);                      // 1,048,576

    // ---- phase A: tokenizer ----
    pack_convw_kernel<<<(512 * 576 + 255) / 256, 256, 0, stream>>>(conv_w, Wct);
    {
        size_t total = (size_t)32768 * 576;
        im2col_kernel<<<(unsigned)((total + 255) / 256), 256, 0, stream>>>(x, Aim);
    }
    wmma_gemm_kernel<_Float16><<<dim3(32768 / 32, 1), 256, 0, stream>>>(
        Aim, Wct, tokh, nullptr, 32768, 512, 576);
    film_kernel<<<(512 * 1024 + 255) / 256, 256, 0, stream>>>(z, film_w, film_b, gb);
    tokenize_kernel<<<512, 256, 0, stream>>>(tokh, conv_b, tok_g, tok_b, gb, h);

    // ---- phase B: pack transformer weights (transposed, f16) ----
    for (int i = 0; i < 6; ++i) {
        pack_transpose_kernel<<<(512 * 1536 + 255) / 256, 256, 0, stream>>>(
            kqv_w + (size_t)i * 512 * 1536, kqvT + (size_t)i * 1536 * 512, 512, 1536);
        pack_transpose_kernel<<<(512 * 512 + 255) / 256, 256, 0, stream>>>(
            proj_w + (size_t)i * 512 * 512, projT + (size_t)i * 512 * 512, 512, 512);
        pack_transpose_kernel<<<(512 * 2048 + 255) / 256, 256, 0, stream>>>(
            mlp_w1 + (size_t)i * 512 * 2048, m1T + (size_t)i * 2048 * 512, 512, 2048);
        pack_transpose_kernel<<<(2048 * 512 + 255) / 256, 256, 0, stream>>>(
            mlp_w2 + (size_t)i * 2048 * 512, m2T + (size_t)i * 512 * 2048, 2048, 512);
    }

    // ---- transformer layers ----
    for (int i = 0; i < 6; ++i) {
        ln_pack_kernel<<<512, 256, 0, stream>>>(h, ln1_g + i * 512, ln1_b + i * 512, xi);
        wmma_gemm_kernel<float><<<dim3(16, 3), 256, 0, stream>>>(
            xi, kqvT + (size_t)i * 1536 * 512, kqv, kqv_b + i * 1536, 512, 1536, 512);
        attention_kernel<<<128, 256, 0, stream>>>(kqv, Yh);
        wmma_gemm_kernel<float><<<dim3(16, 1), 256, 0, stream>>>(
            Yh, projT + (size_t)i * 512 * 512, attnC, proj_b + i * 512, 512, 512, 512);
        residual_kernel<<<1024, 256, 0, stream>>>(h, attnC, rs_attn + i);

        ln_pack_kernel<<<512, 256, 0, stream>>>(h, ln2_g + i * 512, ln2_b + i * 512, xi);
        wmma_gemm_kernel<float><<<dim3(16, 4), 256, 0, stream>>>(
            xi, m1T + (size_t)i * 2048 * 512, m1, mlp_b1 + i * 2048, 512, 2048, 512);
        gelu_kernel<<<(512 * 2048 + 255) / 256, 256, 0, stream>>>(m1, g16, 512 * 2048);
        wmma_gemm_kernel<float><<<dim3(16, 1), 256, 0, stream>>>(
            g16, m2T + (size_t)i * 512 * 2048, m2, mlp_b2 + i * 512, 512, 512, 2048);
        residual_kernel<<<1024, 256, 0, stream>>>(h, m2, rs_mlp + i);
    }

    // ---- score head ----
    head_kernel<<<512, 256, 0, stream>>>(h, head_g, head_b, head_w, head_bs, (float*)d_out);
}
